// ManRNN_38826504356363
// MI455X (gfx1250) — matrix operand order repaired and checked by reading, not compiled, not run on hardware
//
#include <hip/hip_runtime.h>

// ---------------------------------------------------------------------------
// 2-layer tanh RNN for MI455X (gfx1250), bf16 WMMA with f32 accumulation.
//   B=256, T=512, I=128, H=512.
// One workgroup owns a 16-row batch tile and runs the whole recurrence with
// h0/h1 ping-pong state in LDS. 8 wave32s each own 4 N-tiles (16 cols each).
// Weights are pre-converted to bf16 in d_ws (L2-resident, re-read per step).
// x_t is staged cooperatively into LDS (f32 -> bf16) once per step.
// B tiles are software double-buffered so each WMMA's operands were loaded
// one kc-iteration earlier (hides L2 latency at 2 waves/SIMD occupancy).
// Activation uses a fast v_exp_f32-based tanh (state is bf16 anyway).
// ---------------------------------------------------------------------------

typedef __attribute__((ext_vector_type(16))) __bf16 v16bf;
typedef __attribute__((ext_vector_type(8)))  float  v8f;

#define T_LEN  512
#define I_DIM  128
#define H_DIM  512
#define B_TILE 16
#define NT_PW  4   // N-tiles per wave: (H_DIM/16)/8 waves

#define WMMA_BF16(a, b, c) \
  __builtin_amdgcn_wmma_f32_16x16x32_bf16(false, (a), false, (b), (short)0, (c), false, false)

// Fast tanh: tanh(x) = (e^2x - 1)/(e^2x + 1), e^2x = exp2(x * 2*log2(e)).
// One v_exp_f32 + one v_rcp_f32 (both TRANS pipe, co-executes with WMMA).
__device__ __forceinline__ float tanh_fast(float x) {
  float cx = fminf(fmaxf(x, -20.0f), 20.0f);
  float t  = __builtin_amdgcn_exp2f(cx * 2.8853900817779268f);  // 2/ln(2)
  return (t - 1.0f) * __builtin_amdgcn_rcpf(t + 1.0f);
}

// B-matrix tile (32x16): B[k][n] = W[n][k], W row-major with leading dim ldk.
// Contiguous k order used here (not the ISA table order): lane n = lane&15, half-group g = lane>>4 holds K = g*16 + i,
// i = 0..15 -> one contiguous 32-byte run of row n.
__device__ __forceinline__ v16bf load_b_tile(const unsigned short* __restrict__ W,
                                             int ldk, int n, int kc, int g) {
  union { uint4 q[2]; v16bf v; } u;
  const uint4* p = reinterpret_cast<const uint4*>(W + (size_t)n * ldk + kc + g * 16);
  u.q[0] = p[0];
  u.q[1] = p[1];
  return u.v;
}

// A-matrix tile (16x32) from LDS bf16 data, row stride ldk.
// Contiguous k order used here (not the ISA table order): lane m = lane&15; element i holds K = kc + g*16 + i
// (matches load_b_tile)  -> two 16-byte chunks.
__device__ __forceinline__ v16bf load_a_lds(const unsigned short* h, int ldk,
                                            int m, int kc, int g) {
  union { uint4 q[2]; v16bf v; } u;
  const unsigned short* base = h + m * ldk + kc + g * 16;
  u.q[0] = *reinterpret_cast<const uint4*>(base);
  u.q[1] = *reinterpret_cast<const uint4*>(base + 8);
  return u.v;
}

// Double-buffered K-loop: acc[j] += A(16xK) * W(=B, KxN-tiles). ldk(W)==K,
// K is a power of two. B tiles for chunk kc+32 / kc+64 are issued before the
// WMMAs consuming chunk kc / kc+32, so each WMMA waits on loads issued a full
// iteration earlier. The wrapped prefetch on the last chunk is harmless.
template <int K>
__device__ __forceinline__ void gemm_db(const unsigned short* __restrict__ W,
                                        const unsigned short* Asrc,
                                        int mn, int g, const int* n0, v8f* acc) {
  v16bf b0[NT_PW], b1[NT_PW];
#pragma unroll
  for (int j = 0; j < NT_PW; ++j) b0[j] = load_b_tile(W, K, n0[j] + mn, 0, g);
#pragma unroll 1
  for (int kc = 0; kc < K; kc += 64) {
    v16bf a0 = load_a_lds(Asrc, K, mn, kc, g);
#pragma unroll
    for (int j = 0; j < NT_PW; ++j) b1[j] = load_b_tile(W, K, n0[j] + mn, kc + 32, g);
#pragma unroll
    for (int j = 0; j < NT_PW; ++j) acc[j] = WMMA_BF16(a0, b0[j], acc[j]);
    v16bf a1 = load_a_lds(Asrc, K, mn, kc + 32, g);
#pragma unroll
    for (int j = 0; j < NT_PW; ++j)
      b0[j] = load_b_tile(W, K, n0[j] + mn, (kc + 64) & (K - 1), g);
#pragma unroll
    for (int j = 0; j < NT_PW; ++j) acc[j] = WMMA_BF16(a1, b1[j], acc[j]);
  }
}

__global__ void ManRNN_cvt_bf16(const float* __restrict__ src,
                                unsigned short* __restrict__ dst, int n) {
  int i = blockIdx.x * 256 + threadIdx.x;
  if (i < n) {
    __bf16 b = (__bf16)src[i];
    dst[i] = __builtin_bit_cast(unsigned short, b);
  }
}

__global__ __launch_bounds__(256, 1) void ManRNN_main_kernel(
    const float* __restrict__ x,
    const unsigned short* __restrict__ Wxh0,
    const float* __restrict__ bxh0,
    const unsigned short* __restrict__ Whh0,
    const unsigned short* __restrict__ Wxh1,
    const float* __restrict__ bxh1,
    const unsigned short* __restrict__ Whh1,
    const float* __restrict__ fcW,
    const float* __restrict__ fcb,
    float* __restrict__ out) {
  __shared__ unsigned short h0buf[2][B_TILE * H_DIM];   // bf16 bits, ping-pong
  __shared__ unsigned short h1buf[2][B_TILE * H_DIM];
  __shared__ unsigned short xbuf[B_TILE * I_DIM];       // bf16 x_t tile

  const int tid   = threadIdx.x;
  const int wave  = tid >> 5;
  const int lane  = tid & 31;
  const int g     = lane >> 4;   // half-wave group
  const int mn    = lane & 15;   // A row for A-tiles; B/C col index
  const int bBase = blockIdx.x * B_TILE;

  // h(-1) = 0 lives in buffer 0.
  for (int i = tid; i < B_TILE * H_DIM; i += 256) {
    h0buf[0][i] = 0;
    h1buf[0][i] = 0;
  }
  __syncthreads();

  int   n0[NT_PW];
  float bias0[NT_PW], bias1[NT_PW];
#pragma unroll
  for (int j = 0; j < NT_PW; ++j) {
    n0[j]    = (wave * NT_PW + j) * 16;
    bias0[j] = bxh0[n0[j] + mn];
    bias1[j] = bxh1[n0[j] + mn];
  }

  // Cooperative x staging: each thread owns 8 consecutive floats of the
  // 16x128 tile. row = tid/16, col = (tid%16)*8.
  const int xrowi = tid >> 4;
  const int xcoli = (tid & 15) * 8;
  const float* xsrc0 = x + (size_t)(bBase + xrowi) * T_LEN * I_DIM + xcoli;
  unsigned short* xdst = &xbuf[xrowi * I_DIM + xcoli];

  for (int t = 0; t < T_LEN; ++t) {
    const int r = t & 1;       // read (prev) buffer
    const int w = r ^ 1;       // write (new) buffer

    // ---- stage x_t into LDS as bf16 (all 256 threads, 8 elems each) ------
    {
      const float4* p = reinterpret_cast<const float4*>(xsrc0 + (size_t)t * I_DIM);
      float4 f0 = p[0], f1 = p[1];
      union { uint4 q; unsigned short s[8]; } u;
      __bf16 b0 = (__bf16)f0.x, b1 = (__bf16)f0.y, b2 = (__bf16)f0.z, b3 = (__bf16)f0.w;
      __bf16 b4 = (__bf16)f1.x, b5 = (__bf16)f1.y, b6 = (__bf16)f1.z, b7 = (__bf16)f1.w;
      u.s[0] = __builtin_bit_cast(unsigned short, b0);
      u.s[1] = __builtin_bit_cast(unsigned short, b1);
      u.s[2] = __builtin_bit_cast(unsigned short, b2);
      u.s[3] = __builtin_bit_cast(unsigned short, b3);
      u.s[4] = __builtin_bit_cast(unsigned short, b4);
      u.s[5] = __builtin_bit_cast(unsigned short, b5);
      u.s[6] = __builtin_bit_cast(unsigned short, b6);
      u.s[7] = __builtin_bit_cast(unsigned short, b7);
      *reinterpret_cast<uint4*>(xdst) = u.q;
    }
    __syncthreads();

    // ---------------- layer 0: h0 = tanh(x_t Wxh0^T + bxh0 + h0 Whh0^T) ----
    v8f acc[NT_PW];
#pragma unroll
    for (int j = 0; j < NT_PW; ++j)
#pragma unroll
      for (int v = 0; v < 8; ++v) acc[j][v] = bias0[j];

    gemm_db<I_DIM>(Wxh0, xbuf, mn, g, n0, acc);
    gemm_db<H_DIM>(Whh0, h0buf[r], mn, g, n0, acc);

#pragma unroll
    for (int j = 0; j < NT_PW; ++j)
#pragma unroll
      for (int v = 0; v < 8; ++v) {
        __bf16 hb = (__bf16)tanh_fast(acc[j][v]);
        h0buf[w][(g * 8 + v) * H_DIM + n0[j] + mn] = __builtin_bit_cast(unsigned short, hb);
      }
    __syncthreads();  // all of h0(t) visible before layer 1 consumes it

    // ---------------- layer 1: h1 = tanh(h0 Wxh1^T + bxh1 + h1 Whh1^T) -----
#pragma unroll
    for (int j = 0; j < NT_PW; ++j)
#pragma unroll
      for (int v = 0; v < 8; ++v) acc[j][v] = bias1[j];

    gemm_db<H_DIM>(Wxh1, h0buf[w], mn, g, n0, acc);
    gemm_db<H_DIM>(Whh1, h1buf[r], mn, g, n0, acc);

#pragma unroll
    for (int j = 0; j < NT_PW; ++j)
#pragma unroll
      for (int v = 0; v < 8; ++v) {
        __bf16 hb = (__bf16)tanh_fast(acc[j][v]);
        h1buf[w][(g * 8 + v) * H_DIM + n0[j] + mn] = __builtin_bit_cast(unsigned short, hb);
      }
    __syncthreads();  // protect prev buffers (and xbuf) before next step
  }

  // ---------------- final FC: out = h1 fcW^T + fcb (B,1) -------------------
  // T_LEN is even -> final h1 lives in buffer index (T_LEN & 1) == 0.
  if (tid < B_TILE) {
    const unsigned short* h = &h1buf[T_LEN & 1][tid * H_DIM];
    float acc = fcb[0];
    for (int k = 0; k < H_DIM; ++k)
      acc += (float)__builtin_bit_cast(__bf16, h[k]) * fcW[k];
    out[bBase + tid] = acc;
  }
}

extern "C" void kernel_launch(void* const* d_in, const int* in_sizes, int n_in,
                              void* d_out, int out_size, void* d_ws, size_t ws_size,
                              hipStream_t stream) {
  const float* x    = (const float*)d_in[0];
  const float* Wxh0 = (const float*)d_in[1];
  const float* bxh0 = (const float*)d_in[2];
  const float* Whh0 = (const float*)d_in[3];
  const float* Wxh1 = (const float*)d_in[4];
  const float* bxh1 = (const float*)d_in[5];
  const float* Whh1 = (const float*)d_in[6];
  const float* fcW  = (const float*)d_in[7];
  const float* fcb  = (const float*)d_in[8];

  // bf16 weight copies in workspace (~1.7 MB total).
  unsigned short* ws    = (unsigned short*)d_ws;
  unsigned short* Wxh0b = ws;
  unsigned short* Whh0b = Wxh0b + H_DIM * I_DIM;
  unsigned short* Wxh1b = Whh0b + H_DIM * H_DIM;
  unsigned short* Whh1b = Wxh1b + H_DIM * H_DIM;

  const int nXI = H_DIM * I_DIM;
  const int nHH = H_DIM * H_DIM;
  ManRNN_cvt_bf16<<<(nXI + 255) / 256, 256, 0, stream>>>(Wxh0, Wxh0b, nXI);
  ManRNN_cvt_bf16<<<(nHH + 255) / 256, 256, 0, stream>>>(Whh0, Whh0b, nHH);
  ManRNN_cvt_bf16<<<(nHH + 255) / 256, 256, 0, stream>>>(Wxh1, Wxh1b, nHH);
  ManRNN_cvt_bf16<<<(nHH + 255) / 256, 256, 0, stream>>>(Whh1, Whh1b, nHH);

  ManRNN_main_kernel<<<256 / B_TILE, 256, 0, stream>>>(
      x, Wxh0b, bxh0, Whh0b, Wxh1b, bxh1, Whh1b, fcW, fcb, (float*)d_out);
}
